// HGBCN_19146964205949
// MI455X (gfx1250) — compile-verified, matrix operand order checked by reading, not run on hardware
//
#include <hip/hip_runtime.h>

typedef __attribute__((ext_vector_type(16))) _Float16 v16h;
typedef __attribute__((ext_vector_type(8)))  _Float16 v8h;
typedef __attribute__((ext_vector_type(2)))  _Float16 v2h;
typedef __attribute__((ext_vector_type(8)))  float    v8f;

#define DIM 128
#define K2  256   // 2*DIM

// ---------------------------------------------------------------------------
// hardware f32 atomic add (global_atomic_add_f32, no CAS loop)
// ---------------------------------------------------------------------------
__device__ __forceinline__ void atomAddF(float* p, float v) {
    unsafeAtomicAdd(p, v);
}

// ---------------------------------------------------------------------------
// zero-fill (grid-stride)
// ---------------------------------------------------------------------------
__global__ void hg_zero_kernel(float* __restrict__ p, size_t n) {
    size_t i = (size_t)blockIdx.x * blockDim.x + threadIdx.x;
    size_t stride = (size_t)gridDim.x * blockDim.x;
    for (; i < n; i += stride) p[i] = 0.0f;
}

// ---------------------------------------------------------------------------
// stage 1: node_msg[rows[e], :] += vals[e] * item_emb[cols[e], :]
// one wave32 per edge; lane handles dims lane, lane+32, lane+64, lane+96
// ---------------------------------------------------------------------------
__global__ void __launch_bounds__(256) hg_spmm_fwd(
    const float* __restrict__ item_emb, const int* __restrict__ rows,
    const int* __restrict__ cols, const float* __restrict__ vals,
    float* __restrict__ node_msg, int nnz) {
    int e = blockIdx.x * (blockDim.x >> 5) + (threadIdx.x >> 5);
    if (e >= nnz) return;
    int lane = threadIdx.x & 31;
    int r = rows[e];
    int c = cols[e];
    float v = vals[e];
    const float* src = item_emb + (size_t)c * DIM;
    float* dst = node_msg + (size_t)r * DIM;
#pragma unroll
    for (int i = 0; i < 4; ++i) {
        int d = lane + 32 * i;
        atomAddF(dst + d, v * src[d]);
    }
}

// ---------------------------------------------------------------------------
// stage 4: norm_emb[cols[e], :] += vals[e] * msg[rows[e], :]
// ---------------------------------------------------------------------------
__global__ void __launch_bounds__(256) hg_spmm_bwd(
    const float* __restrict__ msg, const int* __restrict__ rows,
    const int* __restrict__ cols, const float* __restrict__ vals,
    float* __restrict__ norm_emb, int nnz) {
    int e = blockIdx.x * (blockDim.x >> 5) + (threadIdx.x >> 5);
    if (e >= nnz) return;
    int lane = threadIdx.x & 31;
    int r = rows[e];
    int c = cols[e];
    float v = vals[e];
    const float* src = msg + (size_t)r * DIM;
    float* dst = norm_emb + (size_t)c * DIM;
#pragma unroll
    for (int i = 0; i < 4; ++i) {
        int d = lane + 32 * i;
        atomAddF(dst + d, v * src[d]);
    }
}

// ---------------------------------------------------------------------------
// pack 8 consecutive f32 -> 8 f16 elements of a v16h (A operand fragments)
// ---------------------------------------------------------------------------
__device__ __forceinline__ void pack8(v16h& a, int base, const float* __restrict__ p) {
    const float4 x = *reinterpret_cast<const float4*>(p);
    const float4 y = *reinterpret_cast<const float4*>(p + 4);
    a[base + 0] = (_Float16)x.x; a[base + 1] = (_Float16)x.y;
    a[base + 2] = (_Float16)x.z; a[base + 3] = (_Float16)x.w;
    a[base + 4] = (_Float16)y.x; a[base + 5] = (_Float16)y.y;
    a[base + 6] = (_Float16)y.z; a[base + 7] = (_Float16)y.w;
}

__device__ __forceinline__ void pack8_mul(v16h& a, int base,
                                          const float* __restrict__ p,
                                          const float* __restrict__ q) {
    const float4 x = *reinterpret_cast<const float4*>(p);
    const float4 y = *reinterpret_cast<const float4*>(p + 4);
    const float4 u = *reinterpret_cast<const float4*>(q);
    const float4 w = *reinterpret_cast<const float4*>(q + 4);
    a[base + 0] = (_Float16)(x.x * u.x); a[base + 1] = (_Float16)(x.y * u.y);
    a[base + 2] = (_Float16)(x.z * u.z); a[base + 3] = (_Float16)(x.w * u.w);
    a[base + 4] = (_Float16)(y.x * w.x); a[base + 5] = (_Float16)(y.y * w.y);
    a[base + 6] = (_Float16)(y.z * w.z); a[base + 7] = (_Float16)(y.w * w.w);
}

// ---------------------------------------------------------------------------
// stage 2+3 fused: msg = [node_msg | node_msg*user_emb] @ W_agg^T + b_agg
//
// One wave computes a full 16(users) x 128(out dims) row-block:
//   - 8 f32 accumulator tiles held in VGPRs (8 x v8f)
//   - per 32-wide K-block: A fragment built ONCE (f32->f16, gate fused for
//     K>=128), then reused by 8 WMMAs (one per 16-wide N tile)
//   - B (= W_agg^T fragments) staged in LDS as f16 once per block; each
//     fragment is two contiguous 16B ds_load_b128 reads.
//
// A-fragment layout (16-bit A 16x32, ISA 7.12.2): lane group g = lane/16,
// row M = lane%16; elements [0..7] hold K = kb*32 + g*8 + j, elements
// [8..15] hold K = kb*32 + 16 + g*8 + j.  B mirrors A with column N = lane%16
// (B[k][n] = W_agg[n][k], W_agg rows contiguous in k).
// D layout (f32 C/D 16x16): VGPR r, lane group g, col n=lane%16 ->
// element (M = r + 8*g, N = n).
// ---------------------------------------------------------------------------
__global__ void __launch_bounds__(128) hg_msg_gemm(
    const float* __restrict__ node_msg, const float* __restrict__ user_emb,
    const float* __restrict__ W_agg, const float* __restrict__ b_agg,
    float* __restrict__ msg_out, int n_users) {
    // W_agg as f16, row-major [DIM][K2] = 64 KB LDS
    __shared__ __align__(16) _Float16 Bh[DIM * K2];

    const int tid = threadIdx.x;

    // cooperative f32 -> f16 conversion of W_agg into LDS (pairs -> b32 store)
    {
        const float2* __restrict__ wf2 = reinterpret_cast<const float2*>(W_agg);
        v2h* __restrict__ bh2 = reinterpret_cast<v2h*>(Bh);
        const int npairs = DIM * K2 / 2;  // 16384
        for (int i = tid; i < npairs; i += blockDim.x) {
            float2 v = wf2[i];
            v2h h;
            h[0] = (_Float16)v.x;
            h[1] = (_Float16)v.y;
            bh2[i] = h;
        }
    }
    __syncthreads();

    int rb = blockIdx.x * (blockDim.x >> 5) + (tid >> 5);  // 16-user row block
    if (rb * 16 >= n_users) return;

    int lane = tid & 31;
    int g  = lane >> 4;   // lane group 0/1
    int ln = lane & 15;   // A row / B column / D column within tile

    const float* arow_nm = node_msg + (size_t)(rb * 16 + ln) * DIM;
    const float* arow_ue = user_emb + (size_t)(rb * 16 + ln) * DIM;

    v8f c[8];
#pragma unroll
    for (int t = 0; t < 8; ++t) c[t] = (v8f){0.f, 0.f, 0.f, 0.f, 0.f, 0.f, 0.f, 0.f};

#pragma unroll
    for (int kb = 0; kb < 8; ++kb) {
        const int k0 = kb * 32 + g * 8;   // K of elements [0..7]
        const int k1 = k0 + 16;           // K of elements [8..15]

        // A fragment: built once, reused across all 8 N tiles
        v16h a;
        if (kb < 4) {
            pack8(a, 0, arow_nm + k0);
            pack8(a, 8, arow_nm + k1);
        } else {
            pack8_mul(a, 0, arow_nm + (k0 - DIM), arow_ue + (k0 - DIM));
            pack8_mul(a, 8, arow_nm + (k1 - DIM), arow_ue + (k1 - DIM));
        }

#pragma unroll
        for (int tn = 0; tn < 8; ++tn) {
            const _Float16* bp = Bh + (size_t)(tn * 16 + ln) * K2;
            v8h lo = *reinterpret_cast<const v8h*>(bp + k0);  // ds_load_b128
            v8h hi = *reinterpret_cast<const v8h*>(bp + k1);  // ds_load_b128
            v16h b = __builtin_shufflevector(lo, hi, 0, 1, 2, 3, 4, 5, 6, 7,
                                             8, 9, 10, 11, 12, 13, 14, 15);
            c[tn] = __builtin_amdgcn_wmma_f32_16x16x32_f16(
                /*neg_a=*/false, a, /*neg_b=*/false, b,
                /*c_mod=*/(short)0, c[tn], /*reuse_a=*/false, /*reuse_b=*/false);
        }
    }

    // store D tiles + bias
    float* outrow = msg_out + (size_t)(rb * 16) * DIM;
#pragma unroll
    for (int tn = 0; tn < 8; ++tn) {
        float bv = b_agg[tn * 16 + ln];
#pragma unroll
        for (int r = 0; r < 8; ++r) {
            int row = r + 8 * g;
            outrow[(size_t)row * DIM + tn * 16 + ln] = c[tn][r] + bv;
        }
    }
}

// ---------------------------------------------------------------------------
// launch
// ---------------------------------------------------------------------------
extern "C" void kernel_launch(void* const* d_in, const int* in_sizes, int n_in,
                              void* d_out, int out_size, void* d_ws, size_t ws_size,
                              hipStream_t stream) {
    const float* user_emb = (const float*)d_in[0];
    const float* item_emb = (const float*)d_in[1];
    const int*   hg_rows  = (const int*)d_in[2];
    const int*   hg_cols  = (const int*)d_in[3];
    const float* hg_vals  = (const float*)d_in[4];
    const float* W_agg    = (const float*)d_in[5];
    const float* b_agg    = (const float*)d_in[6];

    const int n_users = in_sizes[0] / DIM;   // 100000
    const int n_items = in_sizes[1] / DIM;   // 50000
    const int nnz     = in_sizes[2];         // 1600000

    float* out      = (float*)d_out;
    float* norm_emb = out;                              // [n_items, DIM]
    float* msg      = out + (size_t)n_items * DIM;      // [n_users, DIM]
    float* node_msg = (float*)d_ws;                     // [n_users, DIM] scratch

    // zero accumulation buffers (d_ws / d_out are poisoned, not zeroed)
    hg_zero_kernel<<<2048, 256, 0, stream>>>(node_msg, (size_t)n_users * DIM);
    hg_zero_kernel<<<2048, 256, 0, stream>>>(norm_emb, (size_t)n_items * DIM);

    // stage 1: SpMM scatter into users
    int wavesPerBlock = 256 / 32;
    hg_spmm_fwd<<<(nnz + wavesPerBlock - 1) / wavesPerBlock, 256, 0, stream>>>(
        item_emb, hg_rows, hg_cols, hg_vals, node_msg, nnz);

    // stage 2+3: fused gate + linear via WMMA (1 wave = 16 users x 128 dims)
    int rowBlocks = (n_users + 15) / 16;            // 6250
    hg_msg_gemm<<<(rowBlocks + 3) / 4, 128, 0, stream>>>(
        node_msg, user_emb, W_agg, b_agg, msg, n_users);

    // stage 4: transpose SpMM scatter into items
    hg_spmm_bwd<<<(nnz + wavesPerBlock - 1) / wavesPerBlock, 256, 0, stream>>>(
        msg, hg_rows, hg_cols, hg_vals, norm_emb, nnz);
}
